// ResGCN_14800457302100
// MI455X (gfx1250) — matrix operand order repaired and checked by reading, not compiled, not run on hardware
//
#include <hip/hip_runtime.h>
#include <math.h>

typedef __attribute__((ext_vector_type(16))) _Float16 v16h;
typedef __attribute__((ext_vector_type(8)))  float    v8f;
typedef _Float16 HF;

#define HDIM 128
#define GDIM 128
#define CDIM 10
#define EPSV 1e-5f

// ---------------- helpers ----------------
__device__ __forceinline__ void fatomic_add(float* p, float v) {
    unsafeAtomicAdd(p, v);   // lowers to global_atomic_add_f32 on CDNA
}

__global__ void k_zero(float* p, long n) {
    long i = (long)blockIdx.x * blockDim.x + threadIdx.x;
    long s = (long)gridDim.x * blockDim.x;
    for (; i < n; i += s) p[i] = 0.0f;
}

// deg[src] += 1 over all edges (self-loop +1 added in k_dis)
__global__ void k_deg(const int* __restrict__ src, int E, float* deg) {
    int i = blockIdx.x * blockDim.x + threadIdx.x;
    if (i < E) fatomic_add(&deg[src[i]], 1.0f);
}

// dis[i] = 1/sqrt(deg[i] + 1)   (in place; +1 = self loop, always > 0)
__global__ void k_dis(float* d, int N) {
    int i = blockIdx.x * blockDim.x + threadIdx.x;
    if (i < N) d[i] = rsqrtf(d[i] + 1.0f);
}

// per-column sum & sum-of-squares of an M x 128 matrix -> stats[0:128]=sum, [128:256]=sumsq
__global__ void k_colstats(const float* __restrict__ A, int M, float* stats) {
    int col = threadIdx.x & 127;
    int rg  = threadIdx.x >> 7;                // 0..1
    int row = blockIdx.x * 2 + rg;
    int stride = gridDim.x * 2;
    float s = 0.0f, q = 0.0f;
    for (int r = row; r < M; r += stride) {
        float v = A[(long)r * HDIM + col];
        s += v; q += v * v;
    }
    fatomic_add(&stats[col], s);
    fatomic_add(&stats[HDIM + col], q);
}

// Fold BN (train-mode, biased var) into W:  Wf[k][n] = scale_k * W[k][n],
// biasf[n] = sum_k shift_k * W[k][n].   grid = 128 (n), block = 128 (k)
__global__ void k_fold(const float* __restrict__ W, const float* __restrict__ stats,
                       const float* __restrict__ gam, const float* __restrict__ bet,
                       float invM, float* Wf, float* biasf) {
    int n = blockIdx.x, k = threadIdx.x;
    float m   = stats[k] * invM;
    float var = stats[HDIM + k] * invM - m * m;
    float sc  = gam[k] * rsqrtf(var + EPSV);
    float sh  = bet[k] - m * sc;
    float w   = W[k * HDIM + n];
    Wf[k * HDIM + n] = sc * w;
    __shared__ float red[HDIM];
    red[k] = sh * w;
    __syncthreads();
    for (int o = 64; o > 0; o >>= 1) { if (k < o) red[k] += red[k + o]; __syncthreads(); }
    if (k == 0) biasf[n] = red[0];
}

// ---------------- WMMA GEMM:  C = [relu](A[Mx128] @ Wf[128x128] + biasf) ----------------
// fp32 accuracy via f16 hi/lo split: A@W ~= ah@bh + al@bh + ah@bl.
// Block = 256 thr = 8 waves; wave w owns columns 16w..16w+15; block loops 16-row tiles.
template <int RELU>
__global__ __launch_bounds__(256) void k_gemm_wmma(
        const float* __restrict__ A, const float* __restrict__ Wf,
        const float* __restrict__ biasf, float* __restrict__ C, int M) {
    __shared__ float As[16 * 129];                  // stride 129: bank-conflict-free
    const int tid  = threadIdx.x;
    const int wave = tid >> 5;
    const int lane = tid & 31;
    const int lh   = lane >> 4;                     // lane half (0/1)
    const int l15  = lane & 15;
    const int n    = wave * 16 + l15;               // output column for this lane

    // Preload B fragments (hi/lo) for the 4 K-chunks.
    // B k order used here (contiguous, same as A): lanes 0-15 hold K=0..15 (half j -> K=j), lanes 16-31 hold K=16..31.
    v16h bh[4], bl[4];
#pragma unroll
    for (int c = 0; c < 4; ++c) {
        int kb = c * 32 + lh * 16;
#pragma unroll
        for (int i = 0; i < 16; ++i) {
            float w = Wf[(kb + i) * HDIM + n];
            HF hi = (HF)w;
            bh[c][i] = hi;
            bl[c][i] = (HF)(w - (float)hi);
        }
    }
    const float bias = biasf[n];
    const int ntiles = (M + 15) >> 4;

    for (int tile = blockIdx.x; tile < ntiles; tile += gridDim.x) {
        const int r0 = tile << 4;
        for (int idx = tid; idx < 16 * HDIM; idx += blockDim.x) {
            int r = idx >> 7, cc = idx & 127;
            int gr = r0 + r;
            As[r * 129 + cc] = (gr < M) ? A[(long)gr * HDIM + cc] : 0.0f;
        }
        __syncthreads();

        v8f acc = {};
#pragma unroll
        for (int c = 0; c < 4; ++c) {
            // A k order used here (contiguous, same as B): lanes 0-15: K = kb + 0..15; lanes 16-31: K = kb + 16..31.
            const int kofs = c * 32 + (lh << 4);
            const int row  = l15;
            v16h ah, al;
#pragma unroll
            for (int i = 0; i < 16; ++i) {
                int k = kofs + i;
                float a = As[row * 129 + k];
                HF hi = (HF)a;
                ah[i] = hi;
                al[i] = (HF)(a - (float)hi);
            }
            acc = __builtin_amdgcn_wmma_f32_16x16x32_f16(false, ah, false, bh[c], (short)0, acc, false, false);
            acc = __builtin_amdgcn_wmma_f32_16x16x32_f16(false, al, false, bh[c], (short)0, acc, false, false);
            acc = __builtin_amdgcn_wmma_f32_16x16x32_f16(false, ah, false, bl[c], (short)0, acc, false, false);
        }
        // C/D layout: VGPR j -> M = j + 8*laneHalf, N = lane&15
        if (r0 + 16 <= M) {                         // fast path: no per-row guards
            float* base = C + (long)(r0 + (lh << 3)) * HDIM + n;
#pragma unroll
            for (int j = 0; j < 8; ++j) {
                float v = acc[j] + bias;
                if (RELU) v = fmaxf(v, 0.0f);
                base[(long)j * HDIM] = v;
            }
        } else {
#pragma unroll
            for (int j = 0; j < 8; ++j) {
                int mrow = r0 + j + (lh << 3);
                if (mrow < M) {
                    float v = acc[j] + bias;
                    if (RELU) v = fmaxf(v, 0.0f);
                    C[(long)mrow * HDIM + n] = v;
                }
            }
        }
        __syncthreads();
    }
}

// acc[i][f] = t[i][f]*dis[i]^2 + conv_b[f]   (self-loop message + conv bias)
__global__ void k_initacc(const float* __restrict__ t, const float* __restrict__ dis,
                          const float* __restrict__ cb, float* acc, int N) {
    long gid = (long)blockIdx.x * blockDim.x + threadIdx.x;
    if (gid >= (long)N * HDIM) return;
    int i = (int)(gid >> 7), f = (int)(gid & 127);
    float d = dis[i];
    acc[gid] = t[gid] * d * d + cb[f];
}

// one wave per edge, each lane moves 4 features
__global__ void k_scatter(const float* __restrict__ t, const float* __restrict__ dis,
                          const int* __restrict__ src, const int* __restrict__ dst,
                          int E, float* acc) {
    long gid = (long)blockIdx.x * blockDim.x + threadIdx.x;
    int e = (int)(gid >> 5);
    if (e >= E) return;
    int lane = (int)(gid & 31);
    int s = src[e], d = dst[e];
    float nm = dis[s] * dis[d];
    const float4 v = *(const float4*)(t + (long)s * HDIM + lane * 4);
    float* p = acc + (long)d * HDIM + lane * 4;
    fatomic_add(p + 0, v.x * nm);
    fatomic_add(p + 1, v.y * nm);
    fatomic_add(p + 2, v.z * nm);
    fatomic_add(p + 3, v.w * nm);
}

__global__ void k_relu(const float* __restrict__ a, float* h, long n) {
    long i = (long)blockIdx.x * blockDim.x + threadIdx.x;
    if (i < n) h[i] = fmaxf(a[i], 0.0f);
}

__global__ void k_pool(const float* __restrict__ h, const int* __restrict__ batch,
                       float* g, int N) {
    long gid = (long)blockIdx.x * blockDim.x + threadIdx.x;
    if (gid >= (long)N * HDIM) return;
    int i = (int)(gid >> 7), f = (int)(gid & 127);
    fatomic_add(&g[batch[i] * HDIM + f], h[gid]);
}

// Head: BN -> relu(@W_fc+b) -> BN -> @W_cls+b -> log_softmax. One block, 128 threads.
__global__ void k_head(float* gA, float* gB,
                       const float* __restrict__ g1, const float* __restrict__ b1,
                       const float* __restrict__ Wfc, const float* __restrict__ bfc,
                       const float* __restrict__ g2, const float* __restrict__ b2,
                       const float* __restrict__ Wcl, const float* __restrict__ bcl,
                       float* __restrict__ out) {
    const int tid = threadIdx.x;
    // BN1 (per column) : gA -> gB
    {
        int c = tid; float s = 0, q = 0;
        for (int r = 0; r < GDIM; ++r) { float v = gA[r * HDIM + c]; s += v; q += v * v; }
        float m = s / GDIM, var = q / GDIM - m * m;
        float sc = g1[c] * rsqrtf(var + EPSV), sh = b1[c] - m * sc;
        for (int r = 0; r < GDIM; ++r) gB[r * HDIM + c] = gA[r * HDIM + c] * sc + sh;
    }
    __syncthreads();
    // g2 = relu(gB @ Wfc + bfc) -> gA
    {
        int n = tid;
        for (int rc = 0; rc < 16; ++rc) {
            float a[8]; float bb = bfc[n];
#pragma unroll
            for (int j = 0; j < 8; ++j) a[j] = bb;
            for (int k = 0; k < HDIM; ++k) {
                float w = Wfc[k * HDIM + n];
#pragma unroll
                for (int j = 0; j < 8; ++j) a[j] += gB[(rc * 8 + j) * HDIM + k] * w;
            }
#pragma unroll
            for (int j = 0; j < 8; ++j) gA[(rc * 8 + j) * HDIM + n] = fmaxf(a[j], 0.0f);
        }
    }
    __syncthreads();
    // BN2 in place on gA
    {
        int c = tid; float s = 0, q = 0;
        for (int r = 0; r < GDIM; ++r) { float v = gA[r * HDIM + c]; s += v; q += v * v; }
        float m = s / GDIM, var = q / GDIM - m * m;
        float sc = g2[c] * rsqrtf(var + EPSV), sh = b2[c] - m * sc;
        for (int r = 0; r < GDIM; ++r) gA[r * HDIM + c] = gA[r * HDIM + c] * sc + sh;
    }
    __syncthreads();
    // logits + log_softmax, one row per thread
    {
        int r = tid;
        float lg[CDIM];
        for (int j = 0; j < CDIM; ++j) {
            float a = bcl[j];
            for (int k = 0; k < HDIM; ++k) a += gA[r * HDIM + k] * Wcl[k * CDIM + j];
            lg[j] = a;
        }
        float mx = lg[0];
        for (int j = 1; j < CDIM; ++j) mx = fmaxf(mx, lg[j]);
        float se = 0.0f;
        for (int j = 0; j < CDIM; ++j) se += expf(lg[j] - mx);
        float ls = logf(se);
        for (int j = 0; j < CDIM; ++j) out[r * CDIM + j] = lg[j] - mx - ls;
    }
}

// ---------------- launcher ----------------
extern "C" void kernel_launch(void* const* d_in, const int* in_sizes, int n_in,
                              void* d_out, int out_size, void* d_ws, size_t ws_size,
                              hipStream_t stream) {
    const float* x        = (const float*)d_in[0];
    const int*   eidx     = (const int*)d_in[1];
    const int*   batch    = (const int*)d_in[2];
    const float* bnf_g    = (const float*)d_in[3];
    const float* bnf_b    = (const float*)d_in[4];
    const float* W_feat   = (const float*)d_in[5];
    const float* cbn_g    = (const float*)d_in[6];
    const float* cbn_b    = (const float*)d_in[7];
    const float* conv_W   = (const float*)d_in[8];
    const float* conv_b   = (const float*)d_in[9];
    const float* bnfc_g   = (const float*)d_in[10];
    const float* bnfc_b   = (const float*)d_in[11];
    const float* W_fc     = (const float*)d_in[12];
    const float* b_fc     = (const float*)d_in[13];
    const float* bnh_g    = (const float*)d_in[14];
    const float* bnh_b    = (const float*)d_in[15];
    const float* W_cls    = (const float*)d_in[16];
    const float* b_cls    = (const float*)d_in[17];

    const int N = in_sizes[0] / HDIM;
    const int E = in_sizes[1] / 2;
    const int* src = eidx;
    const int* dst = eidx + E;

    float* ws = (float*)d_ws;
    size_t o = 0;
    float* dis   = ws + o; o += (size_t)N;
    float* hbuf  = ws + o; o += (size_t)N * HDIM;
    float* tbuf  = ws + o; o += (size_t)N * HDIM;
    float* accb  = ws + o; o += (size_t)N * HDIM;
    float* Wf    = ws + o; o += HDIM * HDIM;
    float* bf    = ws + o; o += HDIM;
    float* stats = ws + o; o += 2 * HDIM;
    float* gA    = ws + o; o += GDIM * HDIM;
    float* gB    = ws + o; o += GDIM * HDIM;

    const long NH = (long)N * HDIM;
    const int  bEW  = (int)((NH + 255) / 256);          // elementwise over N*128
    const int  bE   = (E + 255) / 256;
    const int  bN   = (N + 255) / 256;
    const int  bSc  = (int)(((long)E * 32 + 255) / 256);
    const float invN = 1.0f / (float)N;

    // ---- GCN normalization ----
    k_zero<<<256, 256, 0, stream>>>(dis, N);
    k_deg<<<bE, 256, 0, stream>>>(src, E, dis);
    k_dis<<<bN, 256, 0, stream>>>(dis, N);

    // ---- feature layer: h = relu(BN(x) @ W_feat) ----
    k_zero<<<1, 256, 0, stream>>>(stats, 2 * HDIM);
    k_colstats<<<400, 256, 0, stream>>>(x, N, stats);
    k_fold<<<HDIM, HDIM, 0, stream>>>(W_feat, stats, bnf_g, bnf_b, invN, Wf, bf);
    k_gemm_wmma<1><<<1040, 256, 0, stream>>>(x, Wf, bf, hbuf, N);

    // ---- 3 GCN conv layers ----
    for (int i = 0; i < 3; ++i) {
        k_zero<<<1, 256, 0, stream>>>(stats, 2 * HDIM);
        k_colstats<<<400, 256, 0, stream>>>(hbuf, N, stats);
        k_fold<<<HDIM, HDIM, 0, stream>>>(conv_W + (size_t)i * HDIM * HDIM, stats,
                                          cbn_g + i * HDIM, cbn_b + i * HDIM, invN, Wf, bf);
        k_gemm_wmma<0><<<1040, 256, 0, stream>>>(hbuf, Wf, bf, tbuf, N);
        k_initacc<<<bEW, 256, 0, stream>>>(tbuf, dis, conv_b + i * HDIM, accb, N);
        k_scatter<<<bSc, 256, 0, stream>>>(tbuf, dis, src, dst, E, accb);
        k_relu<<<bEW, 256, 0, stream>>>(accb, hbuf, NH);
    }

    // ---- pool + head ----
    k_zero<<<64, 256, 0, stream>>>(gA, GDIM * HDIM);
    k_pool<<<bEW, 256, 0, stream>>>(hbuf, batch, gA, N);
    k_head<<<1, 128, 0, stream>>>(gA, gB, bnfc_g, bnfc_b, W_fc, b_fc,
                                  bnh_g, bnh_b, W_cls, b_cls, (float*)d_out);
}
